// DiTBlockFut_67894843015269
// MI455X (gfx1250) — compile-verified, matrix operand order checked by reading, not run on hardware
//
#include <hip/hip_runtime.h>
#include <math.h>

// ---------------------------------------------------------------------------
// Types for CDNA5 WMMA / TDM
// ---------------------------------------------------------------------------
typedef __attribute__((ext_vector_type(16))) __bf16 bf16x16;
typedef __attribute__((ext_vector_type(8)))  float  f32x8;
typedef unsigned int u32x4 __attribute__((ext_vector_type(4)));
typedef int          i32x4 __attribute__((ext_vector_type(4)));
typedef int          i32x8 __attribute__((ext_vector_type(8)));

union Frag { bf16x16 v; uint4 q[2]; };

static __device__ inline f32x8 wmma_bf16(const bf16x16& a, const bf16x16& b, f32x8 c) {
    return __builtin_amdgcn_wmma_f32_16x16x32_bf16(false, a, false, b, (short)0, c, false, false);
}

// native f32->bf16 (lowers to v_cvt_pk_bf16_f32 on gfx1250)
static __device__ inline unsigned short f2bf(float f) {
    __bf16 h = (__bf16)f;
    return __builtin_bit_cast(unsigned short, h);
}
static __device__ inline float bf2f(unsigned short h) {
    return __uint_as_float(((unsigned int)h) << 16);
}

#define BATCH 4
#define SEQ   512
#define DIM   1024
#define NHEAD 16
#define HDIM  64
#define HIDD  4096

// ---------------------------------------------------------------------------
// Tensor Data Mover: DMA a [rows x 64] bf16 tile (row stride K elems) from
// global to LDS with an 8-element pad after every 64 elements (72-el stride).
// D# per CDNA5 ISA ch.8: group0 = {count|flags, lds_addr, global_addr, type=2}
// group1 = {dsize/pad, tensor dims, tile dims, dim0 stride}.
// ---------------------------------------------------------------------------
#if defined(__has_builtin)
#if __has_builtin(__builtin_amdgcn_tensor_load_to_lds)
#define HAVE_TDM 1
#endif
#endif
#ifndef HAVE_TDM
#define HAVE_TDM 0
#endif

#if HAVE_TDM
static __device__ inline void tdm_load_tile_2d(unsigned lds_addr,
                                               const unsigned short* gptr,
                                               int k_elems) {
    unsigned long long ga = (unsigned long long)(const void*)gptr;
    u32x4 g0 = { 1u,                                    // count=1, no gather
                 lds_addr,                              // LDS byte address
                 (unsigned)ga,
                 (unsigned)(ga >> 32) | (2u << 30) };   // addr[56:32] | type=2
    // dw0: data_size=2B (1<<16) | pad_enable (1<<20) | pad_interval=32dw (4<<22)
    //      | pad_amount=4dw (3<<25)
    unsigned dw0 = (1u << 16) | (1u << 20) | (4u << 22) | (3u << 25);
    unsigned td0 = (unsigned)k_elems;                   // tensor_dim0
    unsigned td1 = 128u;                                // tensor_dim1 (= tile rows)
    i32x8 g1 = { (int)dw0,
                 (int)((td0 & 0xFFFFu) << 16),                 // dim0[15:0] @ [31:16]
                 (int)((td0 >> 16) | ((td1 & 0xFFFFu) << 16)), // dim0[31:16] | dim1[15:0]
                 (int)((td1 >> 16) | (64u << 16)),             // dim1[31:16] | tile_dim0=64
                 (int)128,                                     // tile_dim1=128, tile_dim2=0
                 (int)(unsigned)k_elems,                       // tensor_dim0_stride[31:0]
                 0, 0 };
    i32x4 gz = { 0, 0, 0, 0 };
#if __clang_major__ >= 23
    i32x8 gz8 = { 0, 0, 0, 0, 0, 0, 0, 0 };
    __builtin_amdgcn_tensor_load_to_lds(g0, g1, gz, gz, gz8, 0);
#else
    __builtin_amdgcn_tensor_load_to_lds(g0, g1, gz, gz, 0);
#endif
}
static __device__ inline unsigned lds_byte_off(const void* p) {
    return (unsigned)(unsigned long long)p;   // low 32 bits of flat LDS addr = ds offset
}
#endif

// ---------------------------------------------------------------------------
// fp32 -> bf16 weight conversion (vectorized x4)
// ---------------------------------------------------------------------------
__global__ void cvt_bf16_kernel(const float* __restrict__ src,
                                unsigned short* __restrict__ dst, int n4) {
    int i = blockIdx.x * blockDim.x + threadIdx.x;
    if (i >= n4) return;
    float4 f = ((const float4*)src)[i];
    ushort4 o;
    o.x = f2bf(f.x); o.y = f2bf(f.y); o.z = f2bf(f.z); o.w = f2bf(f.w);
    ((ushort4*)dst)[i] = o;
}

// ---------------------------------------------------------------------------
// mod = silu(t_emb) @ w_ada^T + b_ada  -> [B, 6*D] fp32   (one wave / output)
// ---------------------------------------------------------------------------
__global__ __launch_bounds__(256) void mod_kernel(const float* __restrict__ t_emb,
                                                  const float* __restrict__ w_ada,
                                                  const float* __restrict__ b_ada,
                                                  float* __restrict__ mod) {
    int wid  = (blockIdx.x * blockDim.x + threadIdx.x) >> 5;
    int lane = threadIdx.x & 31;
    int b = wid / (6 * DIM);
    int j = wid % (6 * DIM);
    if (b >= BATCH) return;
    const float* te = t_emb + b * DIM;
    const float* w  = w_ada + (size_t)j * DIM;
    float s = 0.f;
    for (int d = lane; d < DIM; d += 32) {
        float t = te[d];
        s += (t / (1.f + __expf(-t))) * w[d];
    }
    #pragma unroll
    for (int m = 16; m; m >>= 1) s += __shfl_xor(s, m, 32);
    if (lane == 0) mod[b * 6 * DIM + j] = s + b_ada[j];
}

// ---------------------------------------------------------------------------
// Fused LayerNorm + adaLN modulate -> bf16 row  (one 256-thread block / row)
// ---------------------------------------------------------------------------
__global__ __launch_bounds__(256) void ln_mod_kernel(const float* __restrict__ x,
                                                     const float* __restrict__ g,
                                                     const float* __restrict__ beta,
                                                     const float* __restrict__ mod,
                                                     int scale_off, int shift_off,
                                                     unsigned short* __restrict__ out) {
    __shared__ float redS[8], redQ[8];
    int row = blockIdx.x;              // 0..2047
    int b   = row >> 9;
    const float* xr = x + (size_t)row * DIM;
    float v[4], s = 0.f, ss = 0.f;
    #pragma unroll
    for (int i = 0; i < 4; i++) {
        v[i] = xr[threadIdx.x + i * 256];
        s += v[i]; ss += v[i] * v[i];
    }
    #pragma unroll
    for (int m = 16; m; m >>= 1) { s += __shfl_xor(s, m, 32); ss += __shfl_xor(ss, m, 32); }
    int wv = threadIdx.x >> 5, lane = threadIdx.x & 31;
    if (lane == 0) { redS[wv] = s; redQ[wv] = ss; }
    __syncthreads();
    float S = 0.f, Q = 0.f;
    #pragma unroll
    for (int i = 0; i < 8; i++) { S += redS[i]; Q += redQ[i]; }
    float mean = S * (1.f / DIM);
    float var  = Q * (1.f / DIM) - mean * mean;
    float inv  = rsqrtf(var + 1e-5f);
    const float* mrow = mod + b * 6 * DIM;
    #pragma unroll
    for (int i = 0; i < 4; i++) {
        int col = threadIdx.x + i * 256;
        float y = (v[i] - mean) * inv * g[col] + beta[col];
        y = y * (1.f + mrow[scale_off + col]) + mrow[shift_off + col];
        out[(size_t)row * DIM + col] = f2bf(y);
    }
}

// ---------------------------------------------------------------------------
// Generic bf16 WMMA GEMM: C = A[M,K] @ W[N,K]^T + bias, templated epilogue.
// Block = 256 threads = 8 waves, tile 128(M) x 128(N).
// Double-buffered LDS, K-step 64; tiles staged by the Tensor Data Mover
// (wave 0 issues 2 descriptors/step, s_wait_tensorcnt + barrier publishes).
// Wave grid 4(M) x 2(N); each wave: 2 M-subtiles x 4 N-subtiles of 16x16.
// ---------------------------------------------------------------------------
#define GMODE_QKV   0
#define GMODE_GELU  1
#define GMODE_RESID 2

template <int MODE>
__global__ __launch_bounds__(256)
void gemm_bf16_kernel(const unsigned short* __restrict__ A,
                      const unsigned short* __restrict__ W,
                      const float* __restrict__ bias,
                      int M, int N, int K,
                      unsigned short* __restrict__ qout,
                      unsigned short* __restrict__ kout,
                      unsigned short* __restrict__ vout,
                      unsigned short* __restrict__ bf_out,
                      const float* __restrict__ resid,
                      const float* __restrict__ mod, int gate_off,
                      float* __restrict__ f_out) {
    __shared__ unsigned short sA[2][128 * 72];   // 128 rows x 64 K, +8 pad (144B stride)
    __shared__ unsigned short sW[2][128 * 72];
    const int tid  = threadIdx.x;
    const int wave = tid >> 5, lane = tid & 31;
    const int bm = blockIdx.x * 128;
    const int bn = blockIdx.y * 128;
    const int wm = (wave & 3) * 32;
    const int wn = (wave >> 2) * 64;
    const int lrow = lane & 15;
    const int lo   = (lane >> 4) * 8;      // K-offset for frags / row offset for C

    f32x8 acc[2][4];
    #pragma unroll
    for (int i = 0; i < 2; i++)
        #pragma unroll
        for (int j = 0; j < 4; j++) acc[i][j] = (f32x8){};

    const int nk = K >> 6;

#if HAVE_TDM
    const unsigned ldsA[2] = { lds_byte_off(&sA[0][0]), lds_byte_off(&sA[1][0]) };
    const unsigned ldsW[2] = { lds_byte_off(&sW[0][0]), lds_byte_off(&sW[1][0]) };
    if (wave == 0) {
        tdm_load_tile_2d(ldsA[0], &A[(size_t)bm * K], K);
        tdm_load_tile_2d(ldsW[0], &W[(size_t)bn * K], K);
    }
#else
    const int rowL  = tid >> 1;            // 0..127
    const int cbase = (tid & 1) * 32;      // 0 or 32
    uint4 pa[4], pw[4];
    auto fetch = [&](int k0) {
        const unsigned short* a0 = &A[(size_t)(bm + rowL) * K + k0 + cbase];
        const unsigned short* w0 = &W[(size_t)(bn + rowL) * K + k0 + cbase];
        #pragma unroll
        for (int i = 0; i < 4; i++) {
            pa[i] = *(const uint4*)(a0 + i * 8);
            pw[i] = *(const uint4*)(w0 + i * 8);
        }
    };
    auto stage = [&](int buf) {
        unsigned short* da = &sA[buf][rowL * 72 + cbase];
        unsigned short* dw = &sW[buf][rowL * 72 + cbase];
        #pragma unroll
        for (int i = 0; i < 4; i++) {
            *(uint4*)(da + i * 8) = pa[i];
            *(uint4*)(dw + i * 8) = pw[i];
        }
    };
    fetch(0);
    stage(0);
    __syncthreads();
#endif

    for (int step = 0; step < nk; ++step) {
        const int p = step & 1;
#if HAVE_TDM
        if (wave == 0) __builtin_amdgcn_s_wait_tensorcnt(0);  // buf p ready
        __syncthreads();                                      // publish to all waves
        if (wave == 0 && step + 1 < nk) {                     // start next tile DMA
            const int k1 = (step + 1) << 6;
            tdm_load_tile_2d(ldsA[p ^ 1], &A[(size_t)bm * K + k1], K);
            tdm_load_tile_2d(ldsW[p ^ 1], &W[(size_t)bn * K + k1], K);
        }
#else
        if (step + 1 < nk) fetch((step + 1) << 6);   // overlap with WMMA below
#endif

        #pragma unroll
        for (int half = 0; half < 2; half++) {
            const int ko = half * 32;
            Frag a[2], bf[4];
            #pragma unroll
            for (int i = 0; i < 2; i++) {
                int rr = (wm + i * 16 + lrow) * 72 + ko;
                a[i].q[0] = *(const uint4*)&sA[p][rr + lo];
                a[i].q[1] = *(const uint4*)&sA[p][rr + 16 + lo];
            }
            #pragma unroll
            for (int j = 0; j < 4; j++) {
                int rr = (wn + j * 16 + lrow) * 72 + ko;
                bf[j].q[0] = *(const uint4*)&sW[p][rr + lo];
                bf[j].q[1] = *(const uint4*)&sW[p][rr + 16 + lo];
            }
            #pragma unroll
            for (int i = 0; i < 2; i++)
                #pragma unroll
                for (int j = 0; j < 4; j++)
                    acc[i][j] = wmma_bf16(a[i].v, bf[j].v, acc[i][j]);
        }
#if !HAVE_TDM
        if (step + 1 < nk) {
            stage(p ^ 1);
            __syncthreads();
        }
#endif
    }

    // Epilogue (C layout: VGPR r -> M = r + (lane>=16?8:0), N = lane&15)
    #pragma unroll
    for (int i = 0; i < 2; i++) {
        #pragma unroll
        for (int j = 0; j < 4; j++) {
            int col = bn + wn + j * 16 + lrow;
            float bv = bias[col];
            #pragma unroll
            for (int r = 0; r < 8; r++) {
                int row = bm + wm + i * 16 + r + lo;
                float v = acc[i][j][r] + bv;
                if (MODE == GMODE_QKV) {
                    int b = row >> 9, n = row & 511;
                    int s = col >> 10, rem = col & 1023;
                    int h = rem >> 6, d = rem & 63;
                    size_t bh = (size_t)(b * NHEAD + h);
                    if (s == 0)      qout[(bh * SEQ + n) * HDIM + d] = f2bf(v * 0.125f);
                    else if (s == 1) kout[(bh * SEQ + n) * HDIM + d] = f2bf(v);
                    else             vout[(bh * HDIM + d) * SEQ + n] = f2bf(v); // transposed
                } else if (MODE == GMODE_GELU) {
                    float gl = 0.5f * v * (1.f + erff(v * 0.70710678118654752f));
                    bf_out[(size_t)row * N + col] = f2bf(gl);
                } else { // GMODE_RESID
                    int b = row >> 9;
                    float gate = mod[b * 6 * DIM + gate_off + col];
                    f_out[(size_t)row * N + col] =
                        resid[(size_t)row * N + col] + gate * v;
                }
            }
        }
    }
}

// ---------------------------------------------------------------------------
// Rel-pos bias: relu(rel@w_rp1^T+b1)@w_rp2^T+b2 -> bias[B,H,N,N] bf16
// One wave = one (b, n, 16-key tile): VALU hidden -> LDS -> WMMA 16x64 @ 64x16
// Output transposed through LDS so global stores are packed b128.
// ---------------------------------------------------------------------------
__global__ __launch_bounds__(128)
void relpos_bias_kernel(const float* __restrict__ rel,
                        const float* __restrict__ wrp1, const float* __restrict__ brp1,
                        const unsigned short* __restrict__ wrp2,
                        const float* __restrict__ brp2,
                        unsigned short* __restrict__ bias_out) {
    __shared__ unsigned short hid[4][16 * 72];        // hidden tile, 144B stride
    __shared__ unsigned short ctile[4][16 * 17];      // 16 keys x 16 heads, stride 17
    int wave = threadIdx.x >> 5, lane = threadIdx.x & 31;
    int gw    = blockIdx.x * 4 + wave;                // 0 .. 65535
    int mtile = gw & 31;
    int n     = (gw >> 5) & 511;
    int b     = gw >> 14;
    unsigned short* h = hid[wave];
    const float* rp = rel + (((size_t)(b * SEQ + n) * SEQ) + mtile * 16) * 2;
    #pragma unroll 4
    for (int t = 0; t < 32; t++) {
        int idx = t * 32 + lane;                      // covers 16 pos x 64 hidden
        int p = idx >> 6, j = idx & 63;
        float r0v = rp[p * 2 + 0], r1v = rp[p * 2 + 1];
        float hv = fmaxf(r0v * wrp1[j * 2 + 0] + r1v * wrp1[j * 2 + 1] + brp1[j], 0.f);
        h[p * 72 + j] = f2bf(hv);
    }
    __syncthreads();
    int lrow = lane & 15, lo = (lane >> 4) * 8;
    Frag a0, a1, b0, b1;
    a0.q[0] = *(const uint4*)&h[lrow * 72 + lo];
    a0.q[1] = *(const uint4*)&h[lrow * 72 + 16 + lo];
    a1.q[0] = *(const uint4*)&h[lrow * 72 + 32 + lo];
    a1.q[1] = *(const uint4*)&h[lrow * 72 + 48 + lo];
    b0.q[0] = *(const uint4*)&wrp2[lrow * 64 + lo];
    b0.q[1] = *(const uint4*)&wrp2[lrow * 64 + 16 + lo];
    b1.q[0] = *(const uint4*)&wrp2[lrow * 64 + 32 + lo];
    b1.q[1] = *(const uint4*)&wrp2[lrow * 64 + 48 + lo];
    f32x8 c = (f32x8){};
    c = wmma_bf16(a0.v, b0.v, c);
    c = wmma_bf16(a1.v, b1.v, c);
    float bb = brp2[lrow];                            // col = head
    unsigned short* sc = ctile[wave];
    #pragma unroll
    for (int r = 0; r < 8; r++) {
        int m = r + lo;
        sc[m * 17 + lrow] = f2bf(c[r] + bb);          // sc[key][head]
    }
    __syncthreads();
    // transposed packed store: lane -> (head = lane>>1, 8-key chunk = lane&1)
    int hh = lane >> 1, koff = (lane & 1) * 8;
    unsigned short u[8];
    #pragma unroll
    for (int j = 0; j < 8; j++) u[j] = sc[(koff + j) * 17 + hh];
    uint4 pk;
    pk.x = (unsigned)u[0] | ((unsigned)u[1] << 16);
    pk.y = (unsigned)u[2] | ((unsigned)u[3] << 16);
    pk.z = (unsigned)u[4] | ((unsigned)u[5] << 16);
    pk.w = (unsigned)u[6] | ((unsigned)u[7] << 16);
    size_t dst = (((size_t)(b * NHEAD + hh) * SEQ + n) * SEQ) + mtile * 16 + koff;
    *(uint4*)&bias_out[dst] = pk;
}

// ---------------------------------------------------------------------------
// Flash-style attention. Block = 4 waves; wave owns 16 queries of one (b,h).
// Streams 64-key chunks: S = q@k^T (+bias, mask), online softmax, O += P@v.
// q:[B,H,N,HD] (pre-scaled), k:[B,H,N,HD], v:[B,H,HD,N], bias:[B,H,N,N] bf16
// ---------------------------------------------------------------------------
__global__ __launch_bounds__(128)
void attn_kernel(const unsigned short* __restrict__ qb,
                 const unsigned short* __restrict__ kb,
                 const unsigned short* __restrict__ vb,
                 const unsigned short* __restrict__ biasb,
                 const int* __restrict__ amask,
                 unsigned short* __restrict__ attn_out) {
    __shared__ unsigned short pt[4][16 * 72];
    int wave = threadIdx.x >> 5, lane = threadIdx.x & 31;
    int bh = blockIdx.x;                 // 0..63
    int b = bh >> 4, hidx = bh & 15;
    int q0 = blockIdx.y * 64 + wave * 16;
    int lrow = lane & 15, lo = (lane >> 4) * 8;
    unsigned short* hloc = pt[wave];

    const unsigned short* qp = qb + ((size_t)bh * SEQ + q0) * HDIM;
    Frag qa[2];
    qa[0].q[0] = *(const uint4*)&qp[lrow * HDIM + lo];
    qa[0].q[1] = *(const uint4*)&qp[lrow * HDIM + 16 + lo];
    qa[1].q[0] = *(const uint4*)&qp[lrow * HDIM + 32 + lo];
    qa[1].q[1] = *(const uint4*)&qp[lrow * HDIM + 48 + lo];

    f32x8 acc[4];
    float mrun[8], lrun[8];
    #pragma unroll
    for (int c = 0; c < 4; c++) acc[c] = (f32x8){};
    #pragma unroll
    for (int r = 0; r < 8; r++) { mrun[r] = -INFINITY; lrun[r] = 0.f; }

    const unsigned short* kbase = kb + (size_t)bh * SEQ * HDIM;
    const unsigned short* vbase = vb + (size_t)bh * HDIM * SEQ;
    const unsigned short* bbase = biasb + (size_t)bh * SEQ * SEQ;   // [q][key]

    for (int kc = 0; kc < SEQ; kc += 64) {
        f32x8 S[4];
        #pragma unroll
        for (int c = 0; c < 4; c++) {
            int key = kc + c * 16 + lrow;
            const unsigned short* kp = kbase + (size_t)key * HDIM;
            Frag kf0, kf1;
            kf0.q[0] = *(const uint4*)&kp[lo];      kf0.q[1] = *(const uint4*)&kp[16 + lo];
            kf1.q[0] = *(const uint4*)&kp[32 + lo]; kf1.q[1] = *(const uint4*)&kp[48 + lo];
            f32x8 s = (f32x8){};
            s = wmma_bf16(qa[0].v, kf0.v, s);
            s = wmma_bf16(qa[1].v, kf1.v, s);
            int mk = amask[b * SEQ + key];
            #pragma unroll
            for (int r = 0; r < 8; r++) {
                int qrow = q0 + r + lo;
                float bv = bf2f(bbase[(size_t)qrow * SEQ + key]);   // coalesced
                s[r] = mk ? (s[r] + bv) : -1e30f;
            }
            S[c] = s;
        }
        // online softmax per query row (row lives on 16 lanes within a half)
        float sc8[8];
        #pragma unroll
        for (int r = 0; r < 8; r++) {
            float mx = fmaxf(fmaxf(S[0][r], S[1][r]), fmaxf(S[2][r], S[3][r]));
            #pragma unroll
            for (int m = 8; m; m >>= 1) mx = fmaxf(mx, __shfl_xor(mx, m, 32));
            float mnew = fmaxf(mrun[r], mx);
            float sc = __expf(mrun[r] - mnew);
            float rs = 0.f;
            #pragma unroll
            for (int c = 0; c < 4; c++) {
                float p = __expf(S[c][r] - mnew);
                S[c][r] = p; rs += p;
            }
            #pragma unroll
            for (int m = 8; m; m >>= 1) rs += __shfl_xor(rs, m, 32);
            lrun[r] = lrun[r] * sc + rs;
            mrun[r] = mnew;
            sc8[r] = sc;
        }
        #pragma unroll
        for (int c = 0; c < 4; c++)
            #pragma unroll
            for (int r = 0; r < 8; r++) acc[c][r] *= sc8[r];

        __syncthreads();
        #pragma unroll
        for (int c = 0; c < 4; c++)
            #pragma unroll
            for (int r = 0; r < 8; r++)
                hloc[(r + lo) * 72 + c * 16 + lrow] = f2bf(S[c][r]);
        __syncthreads();

        Frag pa0, pa1;
        pa0.q[0] = *(const uint4*)&hloc[lrow * 72 + lo];
        pa0.q[1] = *(const uint4*)&hloc[lrow * 72 + 16 + lo];
        pa1.q[0] = *(const uint4*)&hloc[lrow * 72 + 32 + lo];
        pa1.q[1] = *(const uint4*)&hloc[lrow * 72 + 48 + lo];
        #pragma unroll
        for (int c = 0; c < 4; c++) {
            const unsigned short* vp = vbase + (size_t)(c * 16 + lrow) * SEQ + kc;
            Frag vf0, vf1;
            vf0.q[0] = *(const uint4*)&vp[lo];      vf0.q[1] = *(const uint4*)&vp[16 + lo];
            vf1.q[0] = *(const uint4*)&vp[32 + lo]; vf1.q[1] = *(const uint4*)&vp[48 + lo];
            acc[c] = wmma_bf16(pa0.v, vf0.v, acc[c]);
            acc[c] = wmma_bf16(pa1.v, vf1.v, acc[c]);
        }
    }
    // normalize + store [B*N, D] bf16
    #pragma unroll
    for (int r = 0; r < 8; r++) {
        float inv = lrun[r] > 0.f ? 1.f / lrun[r] : 0.f;
        int qrow = q0 + r + lo;
        #pragma unroll
        for (int c = 0; c < 4; c++)
            attn_out[(size_t)(b * SEQ + qrow) * DIM + hidx * HDIM + c * 16 + lrow] =
                f2bf(acc[c][r] * inv);
    }
}

// ---------------------------------------------------------------------------
// Host-side launch
// ---------------------------------------------------------------------------
extern "C" void kernel_launch(void* const* d_in, const int* in_sizes, int n_in,
                              void* d_out, int out_size, void* d_ws, size_t ws_size,
                              hipStream_t stream) {
    const float* x      = (const float*)d_in[0];
    const float* t_emb  = (const float*)d_in[1];
    const float* relpos = (const float*)d_in[2];
    const int*   amask  = (const int*)  d_in[3];
    const float* w_ada  = (const float*)d_in[4];
    const float* b_ada  = (const float*)d_in[5];
    const float* g1     = (const float*)d_in[6];
    const float* beta1  = (const float*)d_in[7];
    const float* g2     = (const float*)d_in[8];
    const float* beta2  = (const float*)d_in[9];
    const float* w_qkv  = (const float*)d_in[10];
    const float* b_qkv  = (const float*)d_in[11];
    const float* w_proj = (const float*)d_in[12];
    const float* b_proj = (const float*)d_in[13];
    const float* w_rp1  = (const float*)d_in[14];
    const float* b_rp1  = (const float*)d_in[15];
    const float* w_rp2  = (const float*)d_in[16];
    const float* b_rp2  = (const float*)d_in[17];
    const float* w_fc1  = (const float*)d_in[18];
    const float* b_fc1  = (const float*)d_in[19];
    const float* w_fc2  = (const float*)d_in[20];
    const float* b_fc2  = (const float*)d_in[21];

    char* ws = (char*)d_ws;
    auto alloc = [&](size_t bytes) -> char* {
        char* p = ws;
        ws += (bytes + 255) & ~(size_t)255;
        return p;
    };
    const int ROWS = BATCH * SEQ;                               // 2048
    float*          modbuf  = (float*)         alloc((size_t)BATCH * 6 * DIM * 4);
    unsigned short* wqkv_b  = (unsigned short*)alloc((size_t)3 * DIM * DIM * 2);
    unsigned short* wproj_b = (unsigned short*)alloc((size_t)DIM * DIM * 2);
    unsigned short* wfc1_b  = (unsigned short*)alloc((size_t)HIDD * DIM * 2);
    unsigned short* wfc2_b  = (unsigned short*)alloc((size_t)DIM * HIDD * 2);
    unsigned short* wrp2_b  = (unsigned short*)alloc((size_t)NHEAD * 64 * 2);
    unsigned short* xnbuf   = (unsigned short*)alloc((size_t)ROWS * DIM * 2);
    unsigned short* qbuf    = (unsigned short*)alloc((size_t)ROWS * DIM * 2);
    unsigned short* kbuf    = (unsigned short*)alloc((size_t)ROWS * DIM * 2);
    unsigned short* vbuf    = (unsigned short*)alloc((size_t)ROWS * DIM * 2);
    unsigned short* biasbuf = (unsigned short*)alloc((size_t)BATCH * NHEAD * SEQ * SEQ * 2);
    unsigned short* attnbuf = (unsigned short*)alloc((size_t)ROWS * DIM * 2);
    unsigned short* hbuf    = (unsigned short*)alloc((size_t)ROWS * HIDD * 2);
    float*          xres    = (float*)         alloc((size_t)ROWS * DIM * 4);
    float*          outp    = (float*)d_out;

    auto cvt = [&](const float* s, unsigned short* d, size_t n) {
        int n4 = (int)(n / 4);
        cvt_bf16_kernel<<<(n4 + 255) / 256, 256, 0, stream>>>(s, d, n4);
    };
    cvt(w_qkv,  wqkv_b,  (size_t)3 * DIM * DIM);
    cvt(w_proj, wproj_b, (size_t)DIM * DIM);
    cvt(w_fc1,  wfc1_b,  (size_t)HIDD * DIM);
    cvt(w_fc2,  wfc2_b,  (size_t)DIM * HIDD);
    cvt(w_rp2,  wrp2_b,  (size_t)NHEAD * 64);

    mod_kernel<<<(BATCH * 6 * DIM) / 8, 256, 0, stream>>>(t_emb, w_ada, b_ada, modbuf);

    // LN1 + modulate (scale_s @ 1024, shift_s @ 0)
    ln_mod_kernel<<<ROWS, 256, 0, stream>>>(x, g1, beta1, modbuf, 1024, 0, xnbuf);

    // QKV GEMM: [2048,1024] x [3072,1024]^T
    gemm_bf16_kernel<GMODE_QKV><<<dim3(ROWS / 128, 3 * DIM / 128), 256, 0, stream>>>(
        xnbuf, wqkv_b, b_qkv, ROWS, 3 * DIM, DIM,
        qbuf, kbuf, vbuf, nullptr, nullptr, nullptr, 0, nullptr);

    // rel-pos bias -> [B,H,N,N] bf16
    relpos_bias_kernel<<<(BATCH * SEQ * 32) / 4, 128, 0, stream>>>(
        relpos, w_rp1, b_rp1, wrp2_b, b_rp2, biasbuf);

    // attention
    attn_kernel<<<dim3(BATCH * NHEAD, SEQ / 64), 128, 0, stream>>>(
        qbuf, kbuf, vbuf, biasbuf, amask, attnbuf);

    // proj GEMM + gated residual (gate_s @ 2048) -> xres fp32
    gemm_bf16_kernel<GMODE_RESID><<<dim3(ROWS / 128, DIM / 128), 256, 0, stream>>>(
        attnbuf, wproj_b, b_proj, ROWS, DIM, DIM,
        nullptr, nullptr, nullptr, nullptr, x, modbuf, 2048, xres);

    // LN2 + modulate (scale_m @ 4096, shift_m @ 3072)
    ln_mod_kernel<<<ROWS, 256, 0, stream>>>(xres, g2, beta2, modbuf, 4096, 3072, xnbuf);

    // fc1 GEMM + GELU -> h bf16
    gemm_bf16_kernel<GMODE_GELU><<<dim3(ROWS / 128, HIDD / 128), 256, 0, stream>>>(
        xnbuf, wfc1_b, b_fc1, ROWS, HIDD, DIM,
        nullptr, nullptr, nullptr, hbuf, nullptr, nullptr, 0, nullptr);

    // fc2 GEMM + gated residual (gate_m @ 5120) -> d_out fp32
    gemm_bf16_kernel<GMODE_RESID><<<dim3(ROWS / 128, DIM / 128), 256, 0, stream>>>(
        hbuf, wfc2_b, b_fc2, ROWS, DIM, HIDD,
        nullptr, nullptr, nullptr, nullptr, xres, modbuf, 5120, outp);
}
